// PointNetPP_72035191488651
// MI455X (gfx1250) — matrix operand order repaired and checked by reading, not compiled, not run on hardware
//
#include <hip/hip_runtime.h>
#include <cstdint>

// ---------------------------------------------------------------------------
// PointNet++ (2x set-abstraction + global head) for MI455X / gfx1250.
// FPS: register-resident distance arrays, wave32 shfl argmax reduction.
// MLP levels 2/3: v_wmma_f32_16x16x32_f16 (f16 in, f32 accumulate).
// Bulk staging: global_load_async_to_lds_b128 (ASYNCcnt path).
// ---------------------------------------------------------------------------

typedef __attribute__((ext_vector_type(16))) _Float16     v16h;
typedef __attribute__((ext_vector_type(8)))  float        v8f;
typedef __attribute__((ext_vector_type(8)))  unsigned int v8u;

#define DEVINL __device__ __forceinline__

// -------- async global -> LDS (16B chunk), tracked by ASYNCcnt ------------
DEVINL void async_copy_b128(unsigned lds_off, unsigned long long gbase,
                            unsigned goff) {
  asm volatile("global_load_async_to_lds_b128 %0, %1, %2"
               :: "v"(lds_off), "v"(goff), "s"(gbase)
               : "memory");
}
DEVINL void wait_async0() { asm volatile("s_wait_asynccnt 0" ::: "memory"); }
DEVINL void wait_ds0()    { asm volatile("s_wait_dscnt 0" ::: "memory"); }
DEVINL unsigned lds_addr_of(const void* p) {
  return (unsigned)(uintptr_t)p;  // LDS address = low 32 bits of the generic pointer
}

// ============================ FPS ==========================================
// One workgroup (1024 threads) per point cloud. Each thread keeps EPT points
// and their running min-distance in registers; per iteration: argmax via
// wave32 shuffle tree + 32-entry LDS reduce, broadcast winner, min-update.
template<int N, int M, int EPT>
__launch_bounds__(1024)
__global__ void fps_kernel(const float* __restrict__ pts_all,
                           float* __restrict__ out_pts) {
  const int b   = blockIdx.x;
  const int tid = threadIdx.x;
  const float* pts = pts_all + (size_t)b * N * 3;

  float px[EPT], py[EPT], pz[EPT], mind[EPT];
  const int base = tid * EPT;
#pragma unroll
  for (int j = 0; j < EPT; ++j) {
    px[j] = pts[(base + j) * 3 + 0];
    py[j] = pts[(base + j) * 3 + 1];
    pz[j] = pts[(base + j) * 3 + 2];
  }

  __shared__ float sbx, sby, sbz;
  __shared__ int   swin;
  __shared__ float rv[32];
  __shared__ int   ri[32];

  if (tid == 0) {
    sbx = px[0]; sby = py[0]; sbz = pz[0];
    out_pts[(size_t)b * M * 3 + 0] = px[0];
    out_pts[(size_t)b * M * 3 + 1] = py[0];
    out_pts[(size_t)b * M * 3 + 2] = pz[0];
  }
  __syncthreads();
  {
    const float cx = sbx, cy = sby, cz = sbz;
#pragma unroll
    for (int j = 0; j < EPT; ++j) {
      const float dx = px[j] - cx, dy = py[j] - cy, dz = pz[j] - cz;
      mind[j] = dx * dx + dy * dy + dz * dz;
    }
  }

  const int wave = tid >> 5, lane = tid & 31;
  for (int i = 1; i < M; ++i) {
    // local argmax (first-max tie-break: lower index wins)
    float bv = mind[0]; int bi = base;
#pragma unroll
    for (int j = 1; j < EPT; ++j)
      if (mind[j] > bv) { bv = mind[j]; bi = base + j; }
    // wave32 reduce
#pragma unroll
    for (int off = 16; off > 0; off >>= 1) {
      const float ov = __shfl_down(bv, off);
      const int   oi = __shfl_down(bi, off);
      if (ov > bv || (ov == bv && oi < bi)) { bv = ov; bi = oi; }
    }
    if (lane == 0) { rv[wave] = bv; ri[wave] = bi; }
    __syncthreads();
    if (wave == 0) {
      float v = rv[lane]; int ix = ri[lane];
#pragma unroll
      for (int off = 16; off > 0; off >>= 1) {
        const float ov = __shfl_down(v, off);
        const int   oi = __shfl_down(ix, off);
        if (ov > v || (ov == v && oi < ix)) { v = ov; ix = oi; }
      }
      if (lane == 0) swin = ix;
    }
    __syncthreads();
    const int win = swin;
    if (tid == win / EPT) {           // owner thread has coords in registers
      const int j = win - base;
      sbx = px[j]; sby = py[j]; sbz = pz[j];
      out_pts[((size_t)b * M + i) * 3 + 0] = px[j];
      out_pts[((size_t)b * M + i) * 3 + 1] = py[j];
      out_pts[((size_t)b * M + i) * 3 + 2] = pz[j];
    }
    __syncthreads();
    const float cx = sbx, cy = sby, cz = sbz;
#pragma unroll
    for (int j = 0; j < EPT; ++j) {
      const float dx = px[j] - cx, dy = py[j] - cy, dz = pz[j] - cz;
      mind[j] = fminf(mind[j], dx * dx + dy * dy + dz * dz);
    }
  }
}

// ============================ KNN (k=4) ====================================
// Point tiles staged straight into LDS with global_load_async_to_lds_b128.
__launch_bounds__(256)
__global__ void knn_kernel(const float* __restrict__ q,
                           const float* __restrict__ pts,
                           int* __restrict__ nbr, int M, int N) {
  const int b  = blockIdx.y;
  const int qi = blockIdx.x * 256 + threadIdx.x;
  const float* P = pts + (size_t)b * N * 3;     // 16B-aligned: N*12 % 16 == 0
  float qx = 0.f, qy = 0.f, qz = 0.f;
  if (qi < M) {
    const float* Q = q + ((size_t)b * M + qi) * 3;
    qx = Q[0]; qy = Q[1]; qz = Q[2];
  }
  float bd0 = 1e30f, bd1 = 1e30f, bd2 = 1e30f, bd3 = 1e30f;
  int   bi0 = 0, bi1 = 0, bi2 = 0, bi3 = 0;

  __shared__ __align__(16) float tile[512 * 3];  // 6 KB per tile
  const unsigned tile_lds = lds_addr_of(tile);
  const unsigned long long pbase = (unsigned long long)(uintptr_t)P;

  for (int t = 0; t < N; t += 512) {
    __syncthreads();           // previous tile fully consumed
    wait_ds0();                // close WAR window vs in-flight ds_loads
    // 512 pts * 12B = 384 async 16B chunks, issued by 256 threads
    for (int c = threadIdx.x; c < 384; c += 256)
      async_copy_b128(tile_lds + (unsigned)c * 16u, pbase,
                      (unsigned)t * 12u + (unsigned)c * 16u);
    wait_async0();
    __syncthreads();
#pragma unroll 4
    for (int j = 0; j < 512; ++j) {
      const float dx = tile[j * 3 + 0] - qx;
      const float dy = tile[j * 3 + 1] - qy;
      const float dz = tile[j * 3 + 2] - qz;
      const float d  = dx * dx + dy * dy + dz * dz;
      const int  id  = t + j;
      if (d < bd3) {                      // strict < : stable like top_k
        if (d < bd0)      { bd3=bd2;bi3=bi2; bd2=bd1;bi2=bi1; bd1=bd0;bi1=bi0; bd0=d;bi0=id; }
        else if (d < bd1) { bd3=bd2;bi3=bi2; bd2=bd1;bi2=bi1; bd1=d;bi1=id; }
        else if (d < bd2) { bd3=bd2;bi3=bi2; bd2=d;bi2=id; }
        else              { bd3=d;bi3=id; }
      }
    }
  }
  if (qi < M) {
    int* o = nbr + ((size_t)b * M + qi) * 4;
    o[0] = bi0; o[1] = bi1; o[2] = bi2; o[3] = bi3;
  }
}

// ====================== Level-1 MLP (K=3 -> VALU) ==========================
__launch_bounds__(256)
__global__ void mlp1_kernel(const float* __restrict__ points,
                            const int* __restrict__ nbr1,
                            const float* __restrict__ w1, const float* __restrict__ b1,
                            const float* __restrict__ w2, const float* __restrict__ b2,
                            const float* __restrict__ w3, const float* __restrict__ b3,
                            _Float16* __restrict__ f1) {
  const int idx = blockIdx.x * 256 + threadIdx.x;  // b*2048 + cluster
  const int b = idx >> 11;
  float acc[16];
#pragma unroll
  for (int o = 0; o < 16; ++o) acc[o] = 0.0f;      // post-ReLU values are >= 0
  const int* nb = nbr1 + (size_t)idx * 4;
#pragma unroll 1
  for (int t = 0; t < 4; ++t) {
    const int pi = nb[t];
    const float* p = points + ((size_t)(b << 13) + pi) * 3;
    const float x0 = p[0], x1 = p[1], x2 = p[2];
    float h1[8];
#pragma unroll
    for (int o = 0; o < 8; ++o) {
      const float s = b1[o] + x0 * w1[o] + x1 * w1[8 + o] + x2 * w1[16 + o];
      h1[o] = s > 0.f ? s : 0.f;
    }
    float h2[16];
#pragma unroll
    for (int o = 0; o < 16; ++o) {
      float s = b2[o];
#pragma unroll
      for (int k = 0; k < 8; ++k) s += h1[k] * w2[k * 16 + o];
      h2[o] = s > 0.f ? s : 0.f;
    }
#pragma unroll
    for (int o = 0; o < 16; ++o) {
      float s = b3[o];
#pragma unroll
      for (int k = 0; k < 16; ++k) s += h2[k] * w3[k * 16 + o];
      s = s > 0.f ? s : 0.f;
      acc[o] = fmaxf(acc[o], s);
    }
  }
  _Float16* out = f1 + (size_t)idx * 16;
#pragma unroll
  for (int o = 0; o < 16; ++o) out[o] = (_Float16)acc[o];
}

// ================== weight prep: f32 [K][N] -> f16 [N][Kpad] ===============
__global__ void transpose_w_kernel(const float* __restrict__ w,
                                   _Float16* __restrict__ wT,
                                   int K, int N, int Kpad) {
  const int i = blockIdx.x * 256 + threadIdx.x;
  if (i >= N * Kpad) return;
  const int n = i / Kpad, k = i % Kpad;
  wT[i] = (k < K) ? (_Float16)w[k * N + n] : (_Float16)0.0f;
}

// ============================ WMMA helpers =================================
// A fragment 16x32 f16: lane m = lane%16; VGPR j holds K pair:
//   K = (j>=4)*16 + (lane>=16 ? 8 : 0) + (j%4)*2
DEVINL v16h load_a_frag(const _Float16* buf, int stride, int kb, int lane) {
  const unsigned int* p = (const unsigned int*)(buf + (lane & 15) * stride);
  const int koff = (lane & 16) ? 8 : 0;
  v8u u;
#pragma unroll
  for (int j = 0; j < 8; ++j) {
    const int k = kb + ((j & 4) << 2) + koff + ((j & 3) << 1);
    u[j] = p[k >> 1];
  }
  return __builtin_bit_cast(v16h, u);
}

// B fragment 32x16 f16 from transposed weights wT[N][Kpad]:
//   column n = lane%16; VGPR j holds K pair:
//   K = (j>=4)*16 + (lane>=16 ? 8 : 0) + (j%4)*2
DEVINL v16h load_b_frag(const _Float16* wT, int Kpad, int n0, int kb, int lane) {
  const unsigned int* p =
      (const unsigned int*)(wT + (size_t)(n0 + (lane & 15)) * Kpad);
  const int koff = (lane & 16) ? 8 : 0;
  v8u u;
#pragma unroll
  for (int j = 0; j < 8; ++j) {
    const int k = kb + ((j & 4) << 2) + koff + ((j & 3) << 1);
    u[j] = p[k >> 1];
  }
  return __builtin_bit_cast(v16h, u);
}

DEVINL v8f bias_acc(const float* bias, int n0, int lane) {
  const float bv = bias[n0 + (lane & 15)];
  v8f c;
#pragma unroll
  for (int r = 0; r < 8; ++r) c[r] = bv;
  return c;
}

// D layout: VGPR r -> row (r or r+8), col n = lane%16 (+tile offset)
DEVINL void store_d_relu(v8f c, _Float16* buf, int stride, int n0, int lane) {
  const int n  = n0 + (lane & 15);
  const int mb = (lane & 16) ? 8 : 0;
#pragma unroll
  for (int r = 0; r < 8; ++r) {
    const float v = c[r];
    buf[(mb + r) * stride + n] = (_Float16)(v > 0.f ? v : 0.f);
  }
}

template<int KBLK, int NT>
DEVINL void mlp_layer(const _Float16* inb, int inS,
                      const _Float16* wT, int Kpad, const float* bias,
                      _Float16* outb, int outS, int lane) {
  for (int nt = 0; nt < NT; ++nt) {
    v8f c = bias_acc(bias, nt * 16, lane);
#pragma unroll
    for (int kb = 0; kb < KBLK; ++kb) {
      v16h a  = load_a_frag(inb, inS, kb * 32, lane);
      v16h bm = load_b_frag(wT, Kpad, nt * 16, kb * 32, lane);
      c = __builtin_amdgcn_wmma_f32_16x16x32_f16(false, a, false, bm,
                                                 (short)0, c, false, false);
    }
    store_d_relu(c, outb, outS, nt * 16, lane);
  }
  wait_ds0();  // LDS RAW before next layer consumes outb
}

// ====================== Level-2 MLP (WMMA) + k-pool ========================
__launch_bounds__(256)
__global__ void mlp2_wmma_kernel(const _Float16* __restrict__ f1,
                                 const int* __restrict__ nbr2,
                                 const _Float16* __restrict__ wT1,
                                 const _Float16* __restrict__ wT2,
                                 const _Float16* __restrict__ wT3,
                                 const float* __restrict__ b1,
                                 const float* __restrict__ b2,
                                 const float* __restrict__ b3,
                                 _Float16* __restrict__ f2) {
  constexpr int S = 72;  // 36 dwords/row -> 16 rows hit distinct LDS banks
  __shared__ __align__(16) _Float16 bufA[8][16 * S];
  __shared__ __align__(16) _Float16 bufB[8][16 * S];
  const int b = blockIdx.y, chunk = blockIdx.x;
  const int wave = threadIdx.x >> 5, lane = threadIdx.x & 31;
  _Float16* A  = bufA[wave];
  _Float16* Bu = bufB[wave];

  for (int t = wave; t < 32; t += 8) {
    const int cl0 = chunk * 128 + t * 4;  // 4 clusters (16 rows) per tile
    if (lane < 16) {                      // gather neighbor features, pad K
      const int cl = cl0 + (lane >> 2), nb = lane & 3;
      const int src = nbr2[(((size_t)b << 9) + cl) * 4 + nb];
      const unsigned int* sp =
          (const unsigned int*)(f1 + (((size_t)b * 2048 + src) << 4));
      unsigned int* dp = (unsigned int*)(A + lane * S);
#pragma unroll
      for (int j = 0; j < 4; ++j) dp[j] = sp[j];
#pragma unroll
      for (int j = 4; j < 8; ++j) dp[j] = 0u;   // zero-pad K=8..15
    }
    wait_ds0();

    mlp_layer<1, 2>(A,  S, wT1, 32, b1, Bu, S, lane);  // 16 -> 32
    mlp_layer<1, 4>(Bu, S, wT2, 32, b2, A,  S, lane);  // 32 -> 64

    // final 64 -> 64, k=4 max-pool straight from the accumulator layout:
    // rows {0..3}=cl0, {4..7}=cl0+1 (lanes<16); {8..11}=cl0+2, {12..15}=cl0+3
#pragma unroll
    for (int nt = 0; nt < 4; ++nt) {
      v8f c = bias_acc(b3, nt * 16, lane);
#pragma unroll
      for (int kb = 0; kb < 2; ++kb) {
        v16h a  = load_a_frag(A, S, kb * 32, lane);
        v16h bm = load_b_frag(wT3, 64, nt * 16, kb * 32, lane);
        c = __builtin_amdgcn_wmma_f32_16x16x32_f16(false, a, false, bm,
                                                   (short)0, c, false, false);
      }
      float m0 = fmaxf(fmaxf(c[0], c[1]), fmaxf(c[2], c[3]));
      float m1 = fmaxf(fmaxf(c[4], c[5]), fmaxf(c[6], c[7]));
      m0 = m0 > 0.f ? m0 : 0.f;
      m1 = m1 > 0.f ? m1 : 0.f;
      const int n  = nt * 16 + (lane & 15);
      const int cl = cl0 + ((lane & 16) ? 2 : 0);
      f2[(((size_t)b << 9) + cl)     * 64 + n] = (_Float16)m0;
      f2[(((size_t)b << 9) + cl + 1) * 64 + n] = (_Float16)m1;
    }
  }
}

// =================== Level-3 MLP (WMMA) + global pool ======================
__launch_bounds__(256)
__global__ void mlp3_wmma_kernel(const _Float16* __restrict__ f2,
                                 const _Float16* __restrict__ wT1,
                                 const _Float16* __restrict__ wT2,
                                 const _Float16* __restrict__ wT3,
                                 const float* __restrict__ b1,
                                 const float* __restrict__ b2,
                                 const float* __restrict__ b3,
                                 float* __restrict__ out) {
  constexpr int S = 264;  // 132 dwords/row -> 16 rows on distinct banks
  __shared__ __align__(16) _Float16 bufA[8][16 * S];
  __shared__ __align__(16) _Float16 bufB[8][16 * S];
  __shared__ float pool[8][256];
  const int b = blockIdx.x;
  const int wave = threadIdx.x >> 5, lane = threadIdx.x & 31;
  _Float16* A  = bufA[wave];
  _Float16* Bu = bufB[wave];
  const unsigned a_lds = lds_addr_of(A);
  const unsigned long long f2base =
      (unsigned long long)(uintptr_t)(f2 + ((size_t)b << 9) * 64);

  float runmax[16];
#pragma unroll
  for (int nt = 0; nt < 16; ++nt) runmax[nt] = 0.0f;  // ReLU -> outputs >= 0

  for (int t = wave; t < 32; t += 8) {   // 32 row-tiles of 16 over 512 rows
    // stage 16 rows x 128B via async copy: 128 chunks, 4 per lane
    wait_ds0();                          // WAR vs previous tile's ds_loads
#pragma unroll
    for (int i = 0; i < 4; ++i) {
      const int cchunk = lane + 32 * i;
      const unsigned row = (unsigned)cchunk >> 3;
      const unsigned off = ((unsigned)cchunk & 7u) * 16u;
      async_copy_b128(a_lds + row * (unsigned)(S * 2) + off, f2base,
                      ((unsigned)t * 16u + row) * 128u + off);
    }
    wait_async0();

    mlp_layer<2, 8>(A,  S, wT1, 64,  b1, Bu, S, lane);   // 64  -> 128
    mlp_layer<4, 16>(Bu, S, wT2, 128, b2, A,  S, lane);  // 128 -> 256

    // final 256 -> 256, pool rows straight out of the accumulators
    for (int nt = 0; nt < 16; ++nt) {
      v8f c = bias_acc(b3, nt * 16, lane);
#pragma unroll
      for (int kb = 0; kb < 8; ++kb) {
        v16h a  = load_a_frag(A, S, kb * 32, lane);
        v16h bm = load_b_frag(wT3, 256, nt * 16, kb * 32, lane);
        c = __builtin_amdgcn_wmma_f32_16x16x32_f16(false, a, false, bm,
                                                   (short)0, c, false, false);
      }
      float m = c[0];
#pragma unroll
      for (int r = 1; r < 8; ++r) m = fmaxf(m, c[r]);
      m = m > 0.f ? m : 0.f;
      m = fmaxf(m, __shfl_xor(m, 16));   // fold rows 0-7 with 8-15
      runmax[nt] = fmaxf(runmax[nt], m);
    }
  }

  if (lane < 16) {
#pragma unroll
    for (int nt = 0; nt < 16; ++nt) pool[wave][nt * 16 + lane] = runmax[nt];
  }
  __syncthreads();
  {
    const int tid = threadIdx.x;
    float m = pool[0][tid];
#pragma unroll
    for (int w = 1; w < 8; ++w) m = fmaxf(m, pool[w][tid]);
    out[b * 256 + tid] = m;
  }
}

// ============================== host side ==================================
extern "C" void kernel_launch(void* const* d_in, const int* in_sizes, int n_in,
                              void* d_out, int out_size, void* d_ws,
                              size_t ws_size, hipStream_t stream) {
  (void)in_sizes; (void)n_in; (void)out_size; (void)ws_size;
  const float* points = (const float*)d_in[0];
  // d_in[1] = batch (int64) implied by layout; unused
  const float* pn1_w1 = (const float*)d_in[2];
  const float* pn1_b1 = (const float*)d_in[3];
  const float* pn1_w2 = (const float*)d_in[4];
  const float* pn1_b2 = (const float*)d_in[5];
  const float* pn1_w3 = (const float*)d_in[6];
  const float* pn1_b3 = (const float*)d_in[7];
  const float* pn2_w1 = (const float*)d_in[8];
  const float* pn2_b1 = (const float*)d_in[9];
  const float* pn2_w2 = (const float*)d_in[10];
  const float* pn2_b2 = (const float*)d_in[11];
  const float* pn2_w3 = (const float*)d_in[12];
  const float* pn2_b3 = (const float*)d_in[13];
  const float* pn3_w1 = (const float*)d_in[14];
  const float* pn3_b1 = (const float*)d_in[15];
  const float* pn3_w2 = (const float*)d_in[16];
  const float* pn3_b2 = (const float*)d_in[17];
  const float* pn3_w3 = (const float*)d_in[18];
  const float* pn3_b3 = (const float*)d_in[19];

  char* w = (char*)d_ws;
  auto carve = [&](size_t bytes) -> char* {
    char* p = w;
    w += (bytes + 255) & ~(size_t)255;
    return p;
  };
  float*    pts1 = (float*)carve((size_t)8 * 2048 * 3 * 4);
  int*      nbr1 = (int*)carve((size_t)8 * 2048 * 4 * 4);
  _Float16* f1   = (_Float16*)carve((size_t)8 * 2048 * 16 * 2);
  float*    pts2 = (float*)carve((size_t)8 * 512 * 3 * 4);
  int*      nbr2 = (int*)carve((size_t)8 * 512 * 4 * 4);
  _Float16* f2   = (_Float16*)carve((size_t)8 * 512 * 64 * 2);
  _Float16* wT21 = (_Float16*)carve((size_t)32 * 32 * 2);
  _Float16* wT22 = (_Float16*)carve((size_t)64 * 32 * 2);
  _Float16* wT23 = (_Float16*)carve((size_t)64 * 64 * 2);
  _Float16* wT31 = (_Float16*)carve((size_t)128 * 64 * 2);
  _Float16* wT32 = (_Float16*)carve((size_t)256 * 128 * 2);
  _Float16* wT33 = (_Float16*)carve((size_t)256 * 256 * 2);

  auto tw = [&](const float* src, _Float16* dst, int K, int N, int Kpad) {
    const int total = N * Kpad;
    transpose_w_kernel<<<(total + 255) / 256, 256, 0, stream>>>(src, dst, K, N,
                                                                Kpad);
  };
  tw(pn2_w1, wT21, 16, 32, 32);
  tw(pn2_w2, wT22, 32, 64, 32);
  tw(pn2_w3, wT23, 64, 64, 64);
  tw(pn3_w1, wT31, 64, 128, 64);
  tw(pn3_w2, wT32, 128, 256, 128);
  tw(pn3_w3, wT33, 256, 256, 256);

  // Level 1
  fps_kernel<8192, 2048, 8><<<8, 1024, 0, stream>>>(points, pts1);
  knn_kernel<<<dim3(8, 8), 256, 0, stream>>>(pts1, points, nbr1, 2048, 8192);
  mlp1_kernel<<<64, 256, 0, stream>>>(points, nbr1, pn1_w1, pn1_b1, pn1_w2,
                                      pn1_b2, pn1_w3, pn1_b3, f1);
  // Level 2
  fps_kernel<2048, 512, 2><<<8, 1024, 0, stream>>>(pts1, pts2);
  knn_kernel<<<dim3(2, 8), 256, 0, stream>>>(pts2, pts1, nbr2, 512, 2048);
  mlp2_wmma_kernel<<<dim3(4, 8), 256, 0, stream>>>(f1, nbr2, wT21, wT22, wT23,
                                                   pn2_b1, pn2_b2, pn2_b3, f2);
  // Level 3 + global pool
  mlp3_wmma_kernel<<<8, 256, 0, stream>>>(f2, wT31, wT32, wT33, pn3_b1, pn3_b2,
                                          pn3_b3, (float*)d_out);
}
